// DynamicGCNLayer_77386720739714
// MI455X (gfx1250) — matrix operand order repaired and checked by reading, not compiled, not run on hardware
//
#include <hip/hip_runtime.h>
#include <hip/hip_bf16.h>

// ---- problem constants ------------------------------------------------------
#define N_PTS 1024
#define DIM   128   // IN_DIM == HID == OUT_DIM

// ---- vector types -----------------------------------------------------------
typedef __attribute__((ext_vector_type(16))) _Float16 h16;
typedef __attribute__((ext_vector_type(8)))  _Float16 h8;
typedef __attribute__((ext_vector_type(16))) short    s16;
typedef __attribute__((ext_vector_type(8)))  float    f8;

union H16u { h16 h; s16 s; };

static __device__ __forceinline__ h16 cat8(h8 lo, h8 hi) {
  h16 a;
#pragma unroll
  for (int i = 0; i < 8; ++i) { a[i] = lo[i]; a[i + 8] = hi[i]; }
  return a;
}

static __device__ __forceinline__ f8 wmma_f16(h16 a, h16 b, f8 c) {
  // v_wmma_f32_16x16x32_f16: D = A*B + C
  return __builtin_amdgcn_wmma_f32_16x16x32_f16(
      false, a, false, b, (short)0, c, false, false);
}

// ---- workspace layout (bytes) ----------------------------------------------
#define XH_OFF    (size_t)0         // x as f16          : 1024*128*2 = 262144
#define W1T_OFF   (size_t)262144    // W1^T f16 [h][d]   : 32768
#define WGT_OFF   (size_t)294912    // Wg^T f16 [o][d]   : 32768
#define XW_OFF    (size_t)327680    // x@Wg f32          : 524288
#define DEG_OFF   (size_t)851968    // deg f32           : 4096
#define ADJH_OFF  (size_t)856064    // adj f16           : 2097152
#define YT_OFF    (size_t)2953216   // (dinv_j*xw)^T f16 : 262144

// ---- prep: f16 casts + weight transposes -----------------------------------
__global__ void prep_kernel(const float* __restrict__ x,
                            const float* __restrict__ W1,
                            const float* __restrict__ Wg,
                            _Float16* __restrict__ xh,
                            _Float16* __restrict__ w1t,
                            _Float16* __restrict__ wgt) {
  int idx = blockIdx.x * 256 + threadIdx.x;
  if (idx < N_PTS * DIM) xh[idx] = (_Float16)x[idx];
  if (idx < DIM * DIM) {
    int h = idx >> 7, d = idx & 127;
    w1t[h * DIM + d] = (_Float16)W1[d * DIM + h];
    wgt[h * DIM + d] = (_Float16)Wg[d * DIM + h];
  }
}

// ---- xw = x @ Wg (WMMA, f32 out) -------------------------------------------
__global__ void xw_kernel(const _Float16* __restrict__ xh,
                          const _Float16* __restrict__ wgt,
                          float* __restrict__ xw) {
  int it = blockIdx.x;              // 64 row tiles
  int nt = threadIdx.x >> 5;        // wave id == n-tile 0..7
  int l  = threadIdx.x & 31;
  int m  = l & 15, hi = l >> 4;
  int i0 = it * 16;
  int aoff = hi * 8, boff = hi * 16;
  const _Float16* arow = xh  + (size_t)(i0 + m) * DIM;
  const _Float16* brow = wgt + (size_t)(nt * 16 + m) * DIM;
  f8 c = {};
#pragma unroll
  for (int ks = 0; ks < 4; ++ks) {
    h8  al = *(const h8 *)(arow + ks * 32 + aoff);
    h8  ah = *(const h8 *)(arow + ks * 32 + aoff + 16);
    h16 b  = cat8(*(const h8*)(brow + ks * 32 + aoff), *(const h8*)(brow + ks * 32 + aoff + 16));
    c = wmma_f16(cat8(al, ah), b, c);
  }
#pragma unroll
  for (int r = 0; r < 8; ++r) {
    int row = i0 + r + 8 * hi;
    xw[(size_t)row * DIM + nt * 16 + m] = c[r];
  }
}

// ---- hot kernel: adj[i][j] = sigmoid(w2 . relu(|x_i-x_j| @ W1 + b1) + b2) --
// block = 128 threads (4 waves, 1 wave/SIMD -> big VGPR budget for B-frags)
// grid  = 64 i-tiles * 16 j-slices; each wave handles 16 pair-groups (16 j's)
// x_j fragments are double-buffered in registers so the L2 load latency of
// iteration j+1 hides under iteration j's 32-WMMA burst (XDL co-executes VALU).
__global__ void __launch_bounds__(128) adj_kernel(
    const _Float16* __restrict__ xh, const _Float16* __restrict__ w1t,
    const float* __restrict__ b1, const float* __restrict__ w2,
    const float* __restrict__ b2, float* __restrict__ adj) {
  __shared__ _Float16 xi[16 * 136];     // padded: 272B row stride (16B aligned)
  int tid = threadIdx.x;
  int it = blockIdx.x >> 4;
  int js = blockIdx.x & 15;
  int i0 = it * 16;

  { // stage x_i tile into LDS (16 rows x 128 halfs)
    int m = tid >> 3, cc = (tid & 7) * 16;
    *(h16*)&xi[m * 136 + cc] = *(const h16*)(xh + (size_t)(i0 + m) * DIM + cc);
  }
  __syncthreads();

  int wave = tid >> 5, l = tid & 31;
  int m = l & 15, hi = l >> 4;
  int aoff = hi * 8, boff = hi * 16;

  // x_i fragments (A-layout) — constant across all j of this wave
  h16 xa[4];
#pragma unroll
  for (int ks = 0; ks < 4; ++ks) {
    h8 lo = *(const h8*)&xi[m * 136 + ks * 32 + aoff];
    h8 hv = *(const h8*)&xi[m * 136 + ks * 32 + aoff + 16];
    xa[ks] = cat8(lo, hv);
  }
  // W1^T fragments kept register-resident: 8 n-tiles x 4 k-steps (256 VGPRs)
  h16 bf[8][4];
#pragma unroll
  for (int nt = 0; nt < 8; ++nt) {
    const _Float16* brow = w1t + (size_t)(nt * 16 + m) * DIM;
#pragma unroll
    for (int ks = 0; ks < 4; ++ks)
      bf[nt][ks] = cat8(*(const h8*)(brow + ks * 32 + aoff), *(const h8*)(brow + ks * 32 + aoff + 16));
  }
  float b1v[8], w2v[8];
#pragma unroll
  for (int nt = 0; nt < 8; ++nt) {
    b1v[nt] = b1[nt * 16 + m];
    w2v[nt] = w2[nt * 16 + m];
  }
  float bias2 = b2[0];

  int jbase = js * 64 + wave * 16;

  // prime the x_j double buffer
  h8 cbl[4], cbh[4];
  {
    const _Float16* xj0 = xh + (size_t)jbase * DIM;
#pragma unroll
    for (int ks = 0; ks < 4; ++ks) {
      cbl[ks] = *(const h8*)(xj0 + ks * 32 + aoff);
      cbh[ks] = *(const h8*)(xj0 + ks * 32 + aoff + 16);
    }
  }

  for (int jj = 0; jj < 16; ++jj) {
    int j = jbase + jj;
    // issue next iteration's x_j loads now; they retire under the WMMA burst
    int jn = (jj < 15) ? j + 1 : j;
    const _Float16* xjn = xh + (size_t)jn * DIM;
    h8 pbl[4], pbh[4];
#pragma unroll
    for (int ks = 0; ks < 4; ++ks) {
      pbl[ks] = *(const h8*)(xjn + ks * 32 + aoff);
      pbh[ks] = *(const h8*)(xjn + ks * 32 + aoff + 16);
    }
    // pull a row further ahead into L0 (global_prefetch_b8)
    {
      int jp = (jj < 12) ? j + 4 : j;
      __builtin_prefetch(xh + (size_t)jp * DIM + aoff, 0, 0);
    }

    // A = |x_i - x_j| in the 16-bit A lane layout (packed f16 sub + abs mask)
    h16 af[4];
#pragma unroll
    for (int ks = 0; ks < 4; ++ks) {
      H16u u;
      u.h = xa[ks] - cat8(cbl[ks], cbh[ks]);
      u.s = u.s & (short)0x7fff;
      af[ks] = u.h;
    }
    float psum[8];
#pragma unroll
    for (int r = 0; r < 8; ++r) psum[r] = 0.f;
#pragma unroll
    for (int nt = 0; nt < 8; ++nt) {
      f8 c = {};
#pragma unroll
      for (int ks = 0; ks < 4; ++ks) c = wmma_f16(af[ks], bf[nt][ks], c);
#pragma unroll
      for (int r = 0; r < 8; ++r) {          // relu(+b1), dot with w2 chunk
        float hv = c[r] + b1v[nt];
        hv = hv > 0.f ? hv : 0.f;
        psum[r] += hv * w2v[nt];
      }
    }
    // reduce over N (the 16 lanes of each half-wave group)
#pragma unroll
    for (int r = 0; r < 8; ++r) {
#pragma unroll
      for (int mk = 1; mk < 16; mk <<= 1)
        psum[r] += __shfl_xor(psum[r], mk, 32);
    }
    if (m == 0) {                            // lanes 0 (M=0..7) and 16 (M=8..15)
#pragma unroll
      for (int r = 0; r < 8; ++r) {
        float t = psum[r] + bias2;
        float a = 1.f / (1.f + __expf(-t));
        adj[(size_t)(i0 + r + 8 * hi) * N_PTS + j] = a;
      }
    }
    // rotate double buffer
#pragma unroll
    for (int ks = 0; ks < 4; ++ks) { cbl[ks] = pbl[ks]; cbh[ks] = pbh[ks]; }
  }
}

// ---- deterministic row-sum for deg -----------------------------------------
__global__ void deg_kernel(const float* __restrict__ adj, float* __restrict__ deg) {
  __shared__ float red[256];
  int i = blockIdx.x;
  float s = 0.f;
  for (int j = threadIdx.x; j < N_PTS; j += 256) s += adj[(size_t)i * N_PTS + j];
  red[threadIdx.x] = s;
  __syncthreads();
  for (int off = 128; off > 0; off >>= 1) {
    if (threadIdx.x < off) red[threadIdx.x] += red[threadIdx.x + off];
    __syncthreads();
  }
  if (threadIdx.x == 0) deg[i] = red[0];
}

// ---- cast adj->f16, build y^T[o][j] = dinv_j * xw[j][o] in f16 -------------
__global__ void conv_kernel(const float* __restrict__ adj,
                            const float* __restrict__ xw,
                            const float* __restrict__ deg,
                            _Float16* __restrict__ adjh,
                            _Float16* __restrict__ yt) {
  int idx = blockIdx.x * 256 + threadIdx.x;
  if (idx < N_PTS * N_PTS) adjh[idx] = (_Float16)adj[idx];
  if (idx < DIM * N_PTS) {
    int o = idx >> 10, j = idx & 1023;
    yt[(size_t)o * N_PTS + j] = (_Float16)(rsqrtf(deg[j]) * xw[(size_t)j * DIM + o]);
  }
}

// ---- out = leaky_relu(dinv_i * (adj_h @ y) + bg) ---------------------------
__global__ void out_kernel(const _Float16* __restrict__ adjh,
                           const _Float16* __restrict__ yt,
                           const float* __restrict__ deg,
                           const float* __restrict__ bg,
                           float* __restrict__ out) {
  int it = blockIdx.x;
  int nt = threadIdx.x >> 5;
  int l  = threadIdx.x & 31;
  int m  = l & 15, hi = l >> 4;
  int i0 = it * 16;
  int aoff = hi * 8, boff = hi * 16;
  const _Float16* arow = adjh + (size_t)(i0 + m) * N_PTS;
  const _Float16* brow = yt   + (size_t)(nt * 16 + m) * N_PTS;
  f8 c = {};
#pragma unroll 4
  for (int ks = 0; ks < 32; ++ks) {
    h8  al = *(const h8 *)(arow + ks * 32 + aoff);
    h8  ah = *(const h8 *)(arow + ks * 32 + aoff + 16);
    h16 b  = cat8(*(const h8*)(brow + ks * 32 + aoff), *(const h8*)(brow + ks * 32 + aoff + 16));
    c = wmma_f16(cat8(al, ah), b, c);
  }
  float bgo = bg[nt * 16 + m];
#pragma unroll
  for (int r = 0; r < 8; ++r) {
    int row = i0 + r + 8 * hi;
    float v = rsqrtf(deg[row]) * c[r] + bgo;
    v = v > 0.f ? v : 0.2f * v;
    out[(size_t)row * DIM + nt * 16 + m] = v;
  }
}

// ---- launcher ---------------------------------------------------------------
extern "C" void kernel_launch(void* const* d_in, const int* in_sizes, int n_in,
                              void* d_out, int out_size, void* d_ws, size_t ws_size,
                              hipStream_t stream) {
  (void)in_sizes; (void)n_in; (void)out_size; (void)ws_size;
  const float* x  = (const float*)d_in[0];
  const float* W1 = (const float*)d_in[1];
  const float* b1 = (const float*)d_in[2];
  const float* W2 = (const float*)d_in[3];   // [128,1] -> vector of 128
  const float* b2 = (const float*)d_in[4];
  const float* Wg = (const float*)d_in[5];
  const float* bg = (const float*)d_in[6];

  float* out_p = (float*)d_out;                      // [1024,128]
  float* adj_p = out_p + (size_t)N_PTS * DIM;        // [1024,1024]

  char* ws = (char*)d_ws;
  _Float16* xh   = (_Float16*)(ws + XH_OFF);
  _Float16* w1t  = (_Float16*)(ws + W1T_OFF);
  _Float16* wgt  = (_Float16*)(ws + WGT_OFF);
  float*    xw   = (float*)   (ws + XW_OFF);
  float*    deg  = (float*)   (ws + DEG_OFF);
  _Float16* adjh = (_Float16*)(ws + ADJH_OFF);
  _Float16* yt   = (_Float16*)(ws + YT_OFF);

  prep_kernel<<<512, 256, 0, stream>>>(x, W1, Wg, xh, w1t, wgt);
  xw_kernel<<<64, 256, 0, stream>>>(xh, wgt, xw);
  adj_kernel<<<64 * 16, 128, 0, stream>>>(xh, w1t, b1, W2, b2, adj_p);
  deg_kernel<<<N_PTS, 256, 0, stream>>>(adj_p, deg);
  conv_kernel<<<4096, 256, 0, stream>>>(adj_p, xw, deg, adjh, yt);
  out_kernel<<<64, 256, 0, stream>>>(adjh, yt, deg, bg, out_p);
}
